// PointNet2Binary_59717225284069
// MI455X (gfx1250) — matrix operand order repaired and checked by reading, not compiled, not run on hardware
//
#include <hip/hip_runtime.h>
#include <hip/hip_bf16.h>
#include <stdint.h>
#include <stddef.h>

// ---------------------------------------------------------------------------
// Types for CDNA5 WMMA (wave32): D(16x16 f32) = A(16x32 bf16) x B(32x16 bf16) + C
// ---------------------------------------------------------------------------
typedef __attribute__((ext_vector_type(16))) __bf16 v16bf;
typedef __attribute__((ext_vector_type(8)))  float  v8f;

__device__ __forceinline__ unsigned int f2bf(float f) {
  union { float f; unsigned int u; } x; x.f = f;
  unsigned int r = x.u + 0x7FFFu + ((x.u >> 16) & 1u);   // round-to-nearest-even
  return r >> 16;
}

// Branch-free bounded load: always loads (clamped index), selects 0 when OOB.
__device__ __forceinline__ float load0(const float* __restrict__ p, int i, int n) {
  int ic = i < n ? i : n - 1;
  float v = p[ic];
  return i < n ? v : 0.0f;
}

// ---------------------------------------------------------------------------
// GEMM: Y[M,N] = X[M,K] * W[N,K]^T + bias[N]   (fp32 in/out, bf16 WMMA core)
// One wave per 32x32 output tile, 4 accumulators (2x2 WMMA tiles, operand reuse).
// LDS holds packed bf16x2 K-pairs so each WMMA operand dword = one ds_load_b32:
//   As16[row 0..31][kpair 0..15], Bs16[col 0..31][kpair 0..15]
// A operand (16-bit A 16x32, ISA 7.12.2): lane m=l&15, half=l>>4,
//   dword j -> kpair (j + 4*half) for j<4, (j + 4 + 4*half) for j>=4
// B operand (32x16): lane n=l&15, dword j -> kpair (j + 4*half) for j<4, (j + 4 + 4*half) for j>=4
// C/D f32 16x16: element r = row (r + 8*half), col l&15
// ---------------------------------------------------------------------------
__global__ __launch_bounds__(32) void gemm_bias_wmma(
    const float* __restrict__ Xm, const float* __restrict__ Wm,
    const float* __restrict__ bias, float* __restrict__ Ym,
    int M, int K, int N)
{
  __shared__ unsigned int As16[32 * 16];
  __shared__ unsigned int Bs16[32 * 16];
  const int lane = threadIdx.x;
  const int half = lane >> 4;
  const int l15  = lane & 15;
  const int tm = blockIdx.x * 32;
  const int tn = blockIdx.y * 32;
  v8f acc00 = {}, acc01 = {}, acc10 = {}, acc11 = {};

  for (int k0 = 0; k0 < K; k0 += 32) {
    // Stage A and B tiles as packed bf16 pairs. 512 pair-slots each, 16/lane.
    // Consecutive lanes touch consecutive pair-columns -> coalesced global loads.
#pragma unroll
    for (int i = 0; i < 16; ++i) {
      int pr  = i * 32 + lane;       // 0..511
      int row = pr >> 4;             // 0..31 (A row / B col)
      int pc  = pr & 15;             // k-pair
      int gk  = k0 + 2 * pc;
      const float* sa = Xm + (size_t)(tm + row) * K;
      float a0 = load0(sa, gk, K);
      float a1 = load0(sa, gk + 1, K);
      As16[pr] = f2bf(a0) | (f2bf(a1) << 16);
      const float* sb = Wm + (size_t)(tn + row) * K;
      float w0 = load0(sb, gk, K);
      float w1 = load0(sb, gk + 1, K);
      Bs16[pr] = f2bf(w0) | (f2bf(w1) << 16);
    }
    __syncthreads();

    union OpVec { unsigned int d[8]; v16bf v; } a0, a1, b0, b1;
#pragma unroll
    for (int j = 0; j < 8; ++j) {
      int ka = (j < 4) ? (j + 4 * half) : (j + 4 + 4 * half);
      a0.d[j] = As16[l15 * 16 + ka];
      a1.d[j] = As16[(l15 + 16) * 16 + ka];
      int kb = (j < 4) ? (j + 4 * half) : (j + 4 + 4 * half);
      b0.d[j] = Bs16[l15 * 16 + kb];
      b1.d[j] = Bs16[(l15 + 16) * 16 + kb];
    }
    acc00 = __builtin_amdgcn_wmma_f32_16x16x32_bf16(false, a0.v, false, b0.v, (short)0, acc00, false, false);
    acc01 = __builtin_amdgcn_wmma_f32_16x16x32_bf16(false, a0.v, false, b1.v, (short)0, acc01, false, false);
    acc10 = __builtin_amdgcn_wmma_f32_16x16x32_bf16(false, a1.v, false, b0.v, (short)0, acc10, false, false);
    acc11 = __builtin_amdgcn_wmma_f32_16x16x32_bf16(false, a1.v, false, b1.v, (short)0, acc11, false, false);
    __syncthreads();
  }

  float bv0 = bias[tn + l15];
  float bv1 = bias[tn + 16 + l15];
#pragma unroll
  for (int r = 0; r < 8; ++r) {
    int m0 = tm + r + 8 * half;
    int m1 = m0 + 16;
    Ym[(size_t)m0 * N + tn + l15]        = acc00[r] + bv0;
    Ym[(size_t)m0 * N + tn + 16 + l15]   = acc01[r] + bv1;
    Ym[(size_t)m1 * N + tn + l15]        = acc10[r] + bv0;
    Ym[(size_t)m1 * N + tn + 16 + l15]   = acc11[r] + bv1;
  }
}

// ---------------------------------------------------------------------------
// Per-channel column stats: sum[c], sumsq[c] over M rows. One block per channel.
// ---------------------------------------------------------------------------
__global__ __launch_bounds__(256) void col_stats(
    const float* __restrict__ Y, float* __restrict__ sum, float* __restrict__ sq,
    int M, int C)
{
  int c = blockIdx.x;
  int t = threadIdx.x;
  float s = 0.f, q = 0.f;
  for (int i = t; i < M; i += 256) {
    float v = Y[(size_t)i * C + c];
    s += v; q += v * v;
  }
  __shared__ float ss[256], qq[256];
  ss[t] = s; qq[t] = q; __syncthreads();
  for (int o = 128; o > 0; o >>= 1) {
    if (t < o) { ss[t] += ss[t + o]; qq[t] += qq[t + o]; }
    __syncthreads();
  }
  if (t == 0) { sum[c] = ss[0]; sq[c] = qq[0]; }
}

// BN (training-mode batch stats, biased var) + ReLU, in place.
__global__ __launch_bounds__(256) void bn_relu_kernel(
    float* __restrict__ Y, const float* __restrict__ sum, const float* __restrict__ sq,
    const float* __restrict__ g, const float* __restrict__ be, int M, int C)
{
  size_t i = (size_t)blockIdx.x * 256 + threadIdx.x;
  if (i >= (size_t)M * C) return;
  int c = (int)(i % C);
  float mean = sum[c] / (float)M;
  float var  = sq[c] / (float)M - mean * mean;
  float v = (Y[i] - mean) * rsqrtf(var + 1e-5f) * g[c] + be[c];
  Y[i] = v > 0.f ? v : 0.f;
}

// ---------------------------------------------------------------------------
// Farthest point sampling: one block per batch, dist array lives in LDS.
// Records current farthest BEFORE update (matches lax.scan in reference).
// ---------------------------------------------------------------------------
__global__ __launch_bounds__(256) void fps_kernel(
    const float* __restrict__ xyz, int* __restrict__ out, int N, int npoint)
{
  int b = blockIdx.x;
  const float* p = xyz + (size_t)b * N * 3;
  __shared__ float dist[8192];
  __shared__ float rv[256];
  __shared__ int   ri[256];
  __shared__ int   sfar;
  int t = threadIdx.x;
  for (int i = t; i < N; i += 256) dist[i] = 1e10f;
  if (t == 0) sfar = 0;
  __syncthreads();
  for (int s = 0; s < npoint; ++s) {
    int far = sfar;
    if (t == 0) out[b * npoint + s] = far;
    float cx = p[far * 3], cy = p[far * 3 + 1], cz = p[far * 3 + 2];
    float best = -1.f; int bi = 0x7fffffff;
    for (int i = t; i < N; i += 256) {
      float dx = p[i * 3] - cx, dy = p[i * 3 + 1] - cy, dz = p[i * 3 + 2] - cz;
      float d = dx * dx + dy * dy + dz * dz;
      float nd = fminf(dist[i], d);
      dist[i] = nd;
      if (nd > best) { best = nd; bi = i; }      // first-max wins within thread
    }
    rv[t] = best; ri[t] = bi; __syncthreads();
    for (int o = 128; o > 0; o >>= 1) {
      if (t < o) {
        if (rv[t + o] > rv[t] || (rv[t + o] == rv[t] && ri[t + o] < ri[t])) {
          rv[t] = rv[t + o]; ri[t] = ri[t + o];
        }
      }
      __syncthreads();
    }
    if (t == 0) sfar = ri[0];
    __syncthreads();
  }
}

// Gather xyz rows by fps index: out[b,s,:] = xyz[b, idx[b,s], :]
__global__ __launch_bounds__(256) void gather_xyz(
    const float* __restrict__ xyz, const int* __restrict__ idx,
    float* __restrict__ out, int B, int N, int S)
{
  size_t i = (size_t)blockIdx.x * 256 + threadIdx.x;
  if (i >= (size_t)B * S * 3) return;
  int c = (int)(i % 3);
  size_t bs = i / 3;
  int b = (int)(bs / S);
  int id = idx[bs];
  out[i] = xyz[((size_t)b * N + id) * 3 + c];
}

// Brute force stable 16-NN (ascending distance, lowest index on ties).
__global__ __launch_bounds__(128) void knn16_kernel(
    const float* __restrict__ q, const float* __restrict__ ref,
    int* __restrict__ idx, int B, int S, int N)
{
  int g = blockIdx.x * 128 + threadIdx.x;
  if (g >= B * S) return;
  int b = g / S;
  const float* Q = q + (size_t)g * 3;
  const float* R = ref + (size_t)b * N * 3;
  float qd[16]; int qi[16];
#pragma unroll
  for (int j = 0; j < 16; ++j) { qd[j] = 3.4e38f; qi[j] = 0; }
  float qx = Q[0], qy = Q[1], qz = Q[2];
  for (int i = 0; i < N; ++i) {
    float dx = R[i * 3] - qx, dy = R[i * 3 + 1] - qy, dz = R[i * 3 + 2] - qz;
    float d = dx * dx + dy * dy + dz * dz;
    if (d < qd[15]) {
      int j = 15;
      while (j > 0 && d < qd[j - 1]) { qd[j] = qd[j - 1]; qi[j] = qi[j - 1]; --j; }
      qd[j] = d; qi[j] = i;
    }
  }
#pragma unroll
  for (int j = 0; j < 16; ++j) idx[(size_t)g * 16 + j] = qi[j];
}

// 3-NN with normalized inverse-distance weights (matches fp interpolation).
__global__ __launch_bounds__(128) void knn3_kernel(
    const float* __restrict__ xyz1, const float* __restrict__ xyz2,
    int* __restrict__ idx3, float* __restrict__ w3, int B, int Nn, int S)
{
  int g = blockIdx.x * 128 + threadIdx.x;
  if (g >= B * Nn) return;
  int b = g / Nn;
  const float* Q = xyz1 + (size_t)g * 3;
  const float* R = xyz2 + (size_t)b * S * 3;
  float qd[3]; int qi[3];
#pragma unroll
  for (int j = 0; j < 3; ++j) { qd[j] = 3.4e38f; qi[j] = 0; }
  float qx = Q[0], qy = Q[1], qz = Q[2];
  for (int i = 0; i < S; ++i) {
    float dx = R[i * 3] - qx, dy = R[i * 3 + 1] - qy, dz = R[i * 3 + 2] - qz;
    float d = dx * dx + dy * dy + dz * dz;
    if (d < qd[2]) {
      int j = 2;
      while (j > 0 && d < qd[j - 1]) { qd[j] = qd[j - 1]; qi[j] = qi[j - 1]; --j; }
      qd[j] = d; qi[j] = i;
    }
  }
  float w0 = 1.f / (qd[0] + 1e-8f);
  float w1 = 1.f / (qd[1] + 1e-8f);
  float w2 = 1.f / (qd[2] + 1e-8f);
  float inv = 1.f / (w0 + w1 + w2);
  idx3[(size_t)g * 3 + 0] = qi[0]; w3[(size_t)g * 3 + 0] = w0 * inv;
  idx3[(size_t)g * 3 + 1] = qi[1]; w3[(size_t)g * 3 + 1] = w1 * inv;
  idx3[(size_t)g * 3 + 2] = qi[2]; w3[(size_t)g * 3 + 2] = w2 * inv;
}

// SA grouping: out[b,s,k, 0:3] = xyz[b,idx]-new_xyz[b,s]; out[..,3:3+Cp] = pts[b,idx]
__global__ __launch_bounds__(256) void group_kernel(
    const float* __restrict__ xyz, const float* __restrict__ new_xyz,
    const float* __restrict__ pts, const int* __restrict__ knn,
    float* __restrict__ out, int B, int S, int refN, int Cp, int Ctot)
{
  size_t i = (size_t)blockIdx.x * 256 + threadIdx.x;
  size_t tot = (size_t)B * S * 16 * Ctot;
  if (i >= tot) return;
  int c = (int)(i % Ctot);
  size_t t = i / Ctot;              // (b*S+s)*16 + k
  size_t bs = t >> 4;
  int b = (int)(bs / S);
  int id = knn[t];
  float v;
  if (c < 3) v = xyz[((size_t)b * refN + id) * 3 + c] - new_xyz[bs * 3 + c];
  else       v = pts[((size_t)b * refN + id) * Cp + (c - 3)];
  out[i] = v;
}

// Max over the 16 grouped neighbors: [BS,16,C] -> [BS,C]
__global__ __launch_bounds__(256) void maxpool16(
    const float* __restrict__ in, float* __restrict__ out, size_t BS, int C)
{
  size_t i = (size_t)blockIdx.x * 256 + threadIdx.x;
  if (i >= BS * C) return;
  int c = (int)(i % C);
  size_t bs = i / C;
  float m = -3.4e38f;
#pragma unroll
  for (int k = 0; k < 16; ++k)
    m = fmaxf(m, in[((bs << 4) + k) * C + c]);
  out[i] = m;
}

// Copy a channel block into a wider concat buffer: dst[row, coff+c] = src[row, c]
__global__ __launch_bounds__(256) void copy_cols(
    const float* __restrict__ src, float* __restrict__ dst,
    size_t rows, int Csrc, int Ctot, int coff)
{
  size_t i = (size_t)blockIdx.x * 256 + threadIdx.x;
  if (i >= rows * Csrc) return;
  int c = (int)(i % Csrc);
  size_t r = i / Csrc;
  dst[r * Ctot + coff + c] = src[i];
}

// 3-NN interpolation gather: dst[b,n, coff+c] = sum_j w_j * pts2[b, idx_j, c]
__global__ __launch_bounds__(256) void interp_gather(
    const int* __restrict__ idx3, const float* __restrict__ w3,
    const float* __restrict__ pts2, float* __restrict__ dst,
    int B, int Nn, int S, int C2, int Ctot, int coff)
{
  size_t i = (size_t)blockIdx.x * 256 + threadIdx.x;
  size_t tot = (size_t)B * Nn * C2;
  if (i >= tot) return;
  int c = (int)(i % C2);
  size_t bn = i / C2;
  int b = (int)(bn / Nn);
  const int*   I = idx3 + bn * 3;
  const float* W = w3 + bn * 3;
  const float* P = pts2 + (size_t)b * S * C2;
  float v = W[0] * P[(size_t)I[0] * C2 + c]
          + W[1] * P[(size_t)I[1] * C2 + c]
          + W[2] * P[(size_t)I[2] * C2 + c];
  dst[bn * Ctot + coff + c] = v;
}

// Final classifier: logits[r, 0:2] = h[r, 0:64] . W2[0:2, 0:64]^T + b2
__global__ __launch_bounds__(256) void linear2_kernel(
    const float* __restrict__ h, const float* __restrict__ W,
    const float* __restrict__ b, float* __restrict__ out, size_t rows)
{
  size_t r = (size_t)blockIdx.x * 256 + threadIdx.x;
  if (r >= rows) return;
  float a0 = b[0], a1 = b[1];
  const float* hr = h + r * 64;
#pragma unroll
  for (int c = 0; c < 64; ++c) {
    float v = hr[c];
    a0 += v * W[c];
    a1 += v * W[64 + c];
  }
  out[r * 2 + 0] = a0;
  out[r * 2 + 1] = a1;
}

// ---------------------------------------------------------------------------
// Host orchestration
// ---------------------------------------------------------------------------
struct ConvBN { const float *W, *b, *g, *be; };

static inline unsigned cdiv(size_t a, unsigned b) { return (unsigned)((a + b - 1) / b); }

static void conv_bn_relu(hipStream_t s, const float* in, float* out, const ConvBN& c,
                         int M, int K, int N, float* sum, float* sq)
{
  dim3 grid(M / 32, N / 32);
  gemm_bias_wmma<<<grid, 32, 0, s>>>(in, c.W, c.b, out, M, K, N);
  col_stats<<<N, 256, 0, s>>>(out, sum, sq, M, N);
  size_t tot = (size_t)M * N;
  bn_relu_kernel<<<cdiv(tot, 256), 256, 0, s>>>(out, sum, sq, c.g, c.be, M, N);
}

extern "C" void kernel_launch(void* const* d_in, const int* in_sizes, int n_in,
                              void* d_out, int out_size, void* d_ws, size_t ws_size,
                              hipStream_t stream)
{
  (void)in_sizes; (void)n_in; (void)out_size; (void)ws_size;
  const int B = 16, N0 = 8192, S1 = 512, S2 = 128, S3 = 32;

  // ---- inputs (setup_inputs insertion order; each conv layer = W,b,g,be) ----
  const float* X = (const float*)d_in[0];          // [16,8192,3]
  int p = 1;
  ConvBN sa1[3], sa2[3], sa3[3], fp3[2], fp2[2], fp1[3], cls1;
  auto cb = [&](ConvBN& c) {
    c.W  = (const float*)d_in[p];
    c.b  = (const float*)d_in[p + 1];
    c.g  = (const float*)d_in[p + 2];
    c.be = (const float*)d_in[p + 3];
    p += 4;
  };
  for (int i = 0; i < 3; ++i) cb(sa1[i]);
  for (int i = 0; i < 3; ++i) cb(sa2[i]);
  for (int i = 0; i < 3; ++i) cb(sa3[i]);
  for (int i = 0; i < 2; ++i) cb(fp3[i]);
  for (int i = 0; i < 2; ++i) cb(fp2[i]);
  for (int i = 0; i < 3; ++i) cb(fp1[i]);
  cb(cls1);
  const float* W2 = (const float*)d_in[p];
  const float* b2 = (const float*)d_in[p + 1];

  // ---- workspace bump allocator ----
  uintptr_t base = (uintptr_t)d_ws;
  size_t off = 0;
  auto allocf = [&](size_t n) -> float* {
    float* r = (float*)(base + off); off += ((n * 4 + 255) / 256) * 256; return r;
  };
  auto alloci = [&](size_t n) -> int* {
    int* r = (int*)(base + off); off += ((n * 4 + 255) / 256) * 256; return r;
  };
  float* actA    = allocf((size_t)131072 * 128);   // 64 MB ping
  float* actB    = allocf((size_t)131072 * 128);   // 64 MB pong
  int*   fpsIdx  = alloci(B * S1);
  int*   knnIdx  = alloci((size_t)B * S1 * 16);
  float* l1_xyz  = allocf((size_t)B * S1 * 3);
  float* l2_xyz  = allocf((size_t)B * S2 * 3);
  float* l3_xyz  = allocf((size_t)B * S3 * 3);
  float* l1_pts  = allocf((size_t)B * S1 * 128);
  float* l2_pts  = allocf((size_t)B * S2 * 256);
  float* l3_pts  = allocf((size_t)B * S3 * 512);
  float* l2_ptsB = allocf((size_t)B * S2 * 256);
  float* l1_ptsB = allocf((size_t)B * S1 * 128);
  int*   idx3    = alloci((size_t)B * N0 * 3);
  float* w3      = allocf((size_t)B * N0 * 3);
  float* sumb    = allocf(512);
  float* sqb     = allocf(512);

  // ================= SA1: 8192 -> 512, chs [3,64,64,128] =================
  fps_kernel<<<B, 256, 0, stream>>>(X, fpsIdx, N0, S1);
  gather_xyz<<<cdiv((size_t)B * S1 * 3, 256), 256, 0, stream>>>(X, fpsIdx, l1_xyz, B, N0, S1);
  knn16_kernel<<<cdiv((size_t)B * S1, 128), 128, 0, stream>>>(l1_xyz, X, knnIdx, B, S1, N0);
  {
    size_t tot = (size_t)B * S1 * 16 * 3;
    group_kernel<<<cdiv(tot, 256), 256, 0, stream>>>(X, l1_xyz, nullptr, knnIdx, actB, B, S1, N0, 0, 3);
  }
  conv_bn_relu(stream, actB, actA, sa1[0], 131072, 3, 64, sumb, sqb);
  conv_bn_relu(stream, actA, actB, sa1[1], 131072, 64, 64, sumb, sqb);
  conv_bn_relu(stream, actB, actA, sa1[2], 131072, 64, 128, sumb, sqb);
  maxpool16<<<cdiv((size_t)B * S1 * 128, 256), 256, 0, stream>>>(actA, l1_pts, (size_t)B * S1, 128);

  // ================= SA2: 512 -> 128, chs [131,128,128,256] =================
  fps_kernel<<<B, 256, 0, stream>>>(l1_xyz, fpsIdx, S1, S2);
  gather_xyz<<<cdiv((size_t)B * S2 * 3, 256), 256, 0, stream>>>(l1_xyz, fpsIdx, l2_xyz, B, S1, S2);
  knn16_kernel<<<cdiv((size_t)B * S2, 128), 128, 0, stream>>>(l2_xyz, l1_xyz, knnIdx, B, S2, S1);
  {
    size_t tot = (size_t)B * S2 * 16 * 131;
    group_kernel<<<cdiv(tot, 256), 256, 0, stream>>>(l1_xyz, l2_xyz, l1_pts, knnIdx, actB, B, S2, S1, 128, 131);
  }
  conv_bn_relu(stream, actB, actA, sa2[0], 32768, 131, 128, sumb, sqb);
  conv_bn_relu(stream, actA, actB, sa2[1], 32768, 128, 128, sumb, sqb);
  conv_bn_relu(stream, actB, actA, sa2[2], 32768, 128, 256, sumb, sqb);
  maxpool16<<<cdiv((size_t)B * S2 * 256, 256), 256, 0, stream>>>(actA, l2_pts, (size_t)B * S2, 256);

  // ================= SA3: 128 -> 32, chs [259,256,256,512] =================
  fps_kernel<<<B, 256, 0, stream>>>(l2_xyz, fpsIdx, S2, S3);
  gather_xyz<<<cdiv((size_t)B * S3 * 3, 256), 256, 0, stream>>>(l2_xyz, fpsIdx, l3_xyz, B, S2, S3);
  knn16_kernel<<<cdiv((size_t)B * S3, 128), 128, 0, stream>>>(l3_xyz, l2_xyz, knnIdx, B, S3, S2);
  {
    size_t tot = (size_t)B * S3 * 16 * 259;
    group_kernel<<<cdiv(tot, 256), 256, 0, stream>>>(l2_xyz, l3_xyz, l2_pts, knnIdx, actB, B, S3, S2, 256, 259);
  }
  conv_bn_relu(stream, actB, actA, sa3[0], 8192, 259, 256, sumb, sqb);
  conv_bn_relu(stream, actA, actB, sa3[1], 8192, 256, 256, sumb, sqb);
  conv_bn_relu(stream, actB, actA, sa3[2], 8192, 256, 512, sumb, sqb);
  maxpool16<<<cdiv((size_t)B * S3 * 512, 256), 256, 0, stream>>>(actA, l3_pts, (size_t)B * S3, 512);

  // ========= FP3: interp l3->l2, concat(l2_pts 256, interp 512)=768 ->256->256
  knn3_kernel<<<cdiv((size_t)B * S2, 128), 128, 0, stream>>>(l2_xyz, l3_xyz, idx3, w3, B, S2, S3);
  copy_cols<<<cdiv((size_t)B * S2 * 256, 256), 256, 0, stream>>>(l2_pts, actB, (size_t)B * S2, 256, 768, 0);
  interp_gather<<<cdiv((size_t)B * S2 * 512, 256), 256, 0, stream>>>(idx3, w3, l3_pts, actB, B, S2, S3, 512, 768, 256);
  conv_bn_relu(stream, actB, actA, fp3[0], 2048, 768, 256, sumb, sqb);
  conv_bn_relu(stream, actA, l2_ptsB, fp3[1], 2048, 256, 256, sumb, sqb);

  // ========= FP2: interp l2->l1, concat(l1_pts 128, interp 256)=384 ->256->128
  knn3_kernel<<<cdiv((size_t)B * S1, 128), 128, 0, stream>>>(l1_xyz, l2_xyz, idx3, w3, B, S1, S2);
  copy_cols<<<cdiv((size_t)B * S1 * 128, 256), 256, 0, stream>>>(l1_pts, actB, (size_t)B * S1, 128, 384, 0);
  interp_gather<<<cdiv((size_t)B * S1 * 256, 256), 256, 0, stream>>>(idx3, w3, l2_ptsB, actB, B, S1, S2, 256, 384, 128);
  conv_bn_relu(stream, actB, actA, fp2[0], 8192, 384, 256, sumb, sqb);
  conv_bn_relu(stream, actA, l1_ptsB, fp2[1], 8192, 256, 128, sumb, sqb);

  // ========= FP1: interp l1->l0 (no points1), chs [128,128,128,64]
  knn3_kernel<<<cdiv((size_t)B * N0, 128), 128, 0, stream>>>(X, l1_xyz, idx3, w3, B, N0, S1);
  interp_gather<<<cdiv((size_t)B * N0 * 128, 256), 256, 0, stream>>>(idx3, w3, l1_ptsB, actB, B, N0, S1, 128, 128, 0);
  conv_bn_relu(stream, actB, actA, fp1[0], 131072, 128, 128, sumb, sqb);
  conv_bn_relu(stream, actA, actB, fp1[1], 131072, 128, 128, sumb, sqb);
  conv_bn_relu(stream, actB, actA, fp1[2], 131072, 128, 64, sumb, sqb);

  // ========= Classifier head =========
  conv_bn_relu(stream, actA, actB, cls1, 131072, 64, 64, sumb, sqb);
  linear2_kernel<<<cdiv((size_t)B * N0, 256), 256, 0, stream>>>(actB, W2, b2, (float*)d_out, (size_t)B * N0);
}
